// GCN__37709812859406
// MI455X (gfx1250) — matrix operand order repaired and checked by reading, not compiled, not run on hardware
//
#include <hip/hip_runtime.h>
#include <hip/hip_bf16.h>

typedef __attribute__((ext_vector_type(16))) _Float16 v16h;
typedef __attribute__((ext_vector_type(8)))  _Float16 v8h;
typedef __attribute__((ext_vector_type(8)))  float    v8f;

#define Nn    100000
#define Ee    1600000
#define NCF   40
#define NCP   48      // padded class count (3 x 16 WMMA tiles)
#define LDA   136     // LDS row stride in halves (128 + 8 pad -> conflict-free)

// ---------------- prep: weights -> f16, transposed [n][k]; pad b2 ----------
__global__ __launch_bounds__(256)
void gcn_prep_kernel(const float* __restrict__ W0, const float* __restrict__ W1,
                     const float* __restrict__ W2, const float* __restrict__ b2,
                     _Float16* __restrict__ Wt0, _Float16* __restrict__ Wt1,
                     _Float16* __restrict__ Wt2, float* __restrict__ b2p) {
  int t = threadIdx.x;
  for (int i = t; i < 128 * 128; i += 256) {
    int n = i >> 7, k = i & 127;
    Wt0[n * 128 + k] = (_Float16)W0[k * 128 + n];
    Wt1[n * 128 + k] = (_Float16)W1[k * 128 + n];
  }
  for (int i = t; i < NCP * 128; i += 256) {
    int n = i >> 7, k = i & 127;
    Wt2[n * 128 + k] = (n < NCF) ? (_Float16)W2[k * NCF + n] : (_Float16)0.0f;
  }
  if (t < NCP) b2p[t] = (t < NCF) ? b2[t] : 0.0f;
}

__global__ void gcn_zero_kernel(float* p) { if (threadIdx.x == 0) p[0] = 0.0f; }

// ---------------- GEMM: out = (relu?)X @ Wt^T + b, written to out0 AND out1 -
// X: Nn x 128 f32. Wt: NT x 128 f16 (pre-transposed: Wt[n][k] = W[k][n]).
// Block = 256 thr = 8 wave32. Each wave owns one 16-col tile; its B fragments
// (4 K-steps) live in VGPRs for the whole kernel. Per 16-row stripe: stage A
// f32->f16 into LDS (fused ReLU), then 4x v_wmma_f32_16x16x32_f16.
// NT (== output stride) and RELU are compile-time so the epilogue stores use
// immediate offsets and layer-0 staging has no select overhead.
template<int NT, bool RELU>
__global__ __launch_bounds__(256)
void gcn_gemm_kernel(const float* __restrict__ X,
                     const _Float16* __restrict__ Wt, const float* __restrict__ bias,
                     float* __restrict__ out0, float* __restrict__ out1) {
  __shared__ _Float16 Als[16 * LDA];

  const int t    = threadIdx.x;
  const int lane = t & 31;
  const int wave = t >> 5;
  const int m    = lane & 15;     // A row / B column within tile
  const int hs   = lane >> 4;     // lane-half select
  constexpr int nTiles = NT >> 4;
  const bool active = wave < nTiles;
  const int col = 16 * wave + m;

  // ---- hoist B fragments + bias into registers (constant over stripes) ----
  // B k order used here (contiguous, same as A): lane half h holds K = kb+16h .. kb+16h+15
  v16h bfrag[4];
  float bv = 0.0f;
  if (active) {
    const _Float16* wrow = Wt + (size_t)col * 128;
    #pragma unroll
    for (int ks = 0; ks < 4; ++ks) {
      v8h b0 = *(const v8h*)(wrow + 32 * ks + 16 * hs);
      v8h b1 = *(const v8h*)(wrow + 32 * ks + 16 * hs + 8);
      #pragma unroll
      for (int i = 0; i < 8; ++i) { bfrag[ks][i] = b0[i]; bfrag[ks][8 + i] = b1[i]; }
    }
    bv = bias[col];
  }

  const int nStripes = Nn / 16;
  const int arow = t >> 4, aseg = t & 15;
  const float* asrc = X + ((size_t)blockIdx.x * 16 + arow) * 128 + aseg * 8;
  const size_t astep = (size_t)gridDim.x * 16 * 128;

  float* p0 = out0 + (size_t)(blockIdx.x * 16 + 8 * hs) * NT + col;
  float* p1 = out1 + (size_t)(blockIdx.x * 16 + 8 * hs) * NT + col;
  const size_t ostep = (size_t)gridDim.x * 16 * NT;

  for (int stripe = blockIdx.x; stripe < nStripes; stripe += gridDim.x) {
    // cooperative stage of A: 16 rows x 128 f32 -> f16 (fused ReLU)
    v8h h8;
    #pragma unroll
    for (int i = 0; i < 8; ++i) {
      float f = asrc[i];
      if constexpr (RELU) f = fmaxf(f, 0.0f);
      h8[i] = (_Float16)f;
    }
    *(v8h*)&Als[arow * LDA + aseg * 8] = h8;
    __syncthreads();

    if (active) {
      v8f acc = {};
      #pragma unroll
      for (int ks = 0; ks < 4; ++ks) {
        // A fragment (contiguous k order, same as B): lane half h holds K = kb+16h .. kb+16h+15
        v8h a0 = *(const v8h*)&Als[m * LDA + 32 * ks + 16 * hs];
        v8h a1 = *(const v8h*)&Als[m * LDA + 32 * ks + 16 * hs + 8];
        v16h a;
        #pragma unroll
        for (int i = 0; i < 8; ++i) { a[i] = a0[i]; a[8 + i] = a1[i]; }
        acc = __builtin_amdgcn_wmma_f32_16x16x32_f16(
            false, a, false, bfrag[ks], (short)0, acc, false, false);
      }
      // C/D layout: lane N = lane%16, rows r + 8*(lane/16); compile-time
      // r*NT offsets -> stores with immediate offsets off a single base.
      #pragma unroll
      for (int r = 0; r < 8; ++r) {
        float o = acc[r] + bv;
        p0[(size_t)r * NT] = o;   // gather source
        p1[(size_t)r * NT] = o;   // residual / aggregation destination
      }
    }
    __syncthreads();
    asrc += astep; p0 += ostep; p1 += ostep;
  }
}

// ---------------- sparse aggregation: G[dst] += val * H[src] ---------------
// One wave per edge, 4 edges per wave; H fits in L2 (51 MB < 192 MB) so the
// gather runs at L2 bandwidth; prefetch next edge's row to hide latency.
template<int F>
__global__ __launch_bounds__(256)
void gcn_edge_kernel(const int* __restrict__ src, const int* __restrict__ dst,
                     const float* __restrict__ val,
                     const float* __restrict__ H, float* __restrict__ G, int nE) {
  const int wave = threadIdx.x >> 5;
  const int lane = threadIdx.x & 31;
  int e = (blockIdx.x * 8 + wave) * 4;
  const int eEnd = (e + 4 < nE) ? e + 4 : nE;
  for (; e < eEnd; ++e) {
    if (e + 1 < eEnd)   // global_prefetch_b8 of the next gather row
      __builtin_prefetch(H + (size_t)src[e + 1] * F + lane * (F / 32), 0, 0);
    const int s = src[e], d = dst[e];
    const float v = val[e];
    const float* hrow = H + (size_t)s * F;
    float* grow = G + (size_t)d * F;
    if constexpr (F == 128) {
      float4 hv = *(const float4*)(hrow + lane * 4);
      float* g = grow + lane * 4;
      unsafeAtomicAdd(g + 0, v * hv.x);
      unsafeAtomicAdd(g + 1, v * hv.y);
      unsafeAtomicAdd(g + 2, v * hv.z);
      unsafeAtomicAdd(g + 3, v * hv.w);
    } else {  // F == 48
      if (lane < 24) {
        float2 hv = *(const float2*)(hrow + lane * 2);
        float* g = grow + lane * 2;
        unsafeAtomicAdd(g + 0, v * hv.x);
        unsafeAtomicAdd(g + 1, v * hv.y);
      }
    }
  }
}

// ---------------- cross-entropy (sum) over 40 classes ----------------------
__global__ __launch_bounds__(256)
void gcn_loss_kernel(const float* __restrict__ logits, const int* __restrict__ labels,
                     float* __restrict__ out, int n) {
  __shared__ float red[256];
  int i = blockIdx.x * 256 + threadIdx.x;
  float part = 0.0f;
  if (i < n) {
    const float* l = logits + (size_t)i * NCP;
    float mx = -3.4e38f;
    #pragma unroll
    for (int j = 0; j < NCF; ++j) mx = fmaxf(mx, l[j]);
    float sum = 0.0f;
    #pragma unroll
    for (int j = 0; j < NCF; ++j) sum += __expf(l[j] - mx);
    int lb = labels[i];
    part = -(l[lb] - mx - __logf(sum));
  }
  red[threadIdx.x] = part;
  __syncthreads();
  for (int s = 128; s > 0; s >>= 1) {
    if (threadIdx.x < s) red[threadIdx.x] += red[threadIdx.x + s];
    __syncthreads();
  }
  if (threadIdx.x == 0) unsafeAtomicAdd(out, red[0]);
}

// ---------------------------------------------------------------------------
extern "C" void kernel_launch(void* const* d_in, const int* in_sizes, int n_in,
                              void* d_out, int out_size, void* d_ws, size_t ws_size,
                              hipStream_t stream) {
  (void)in_sizes; (void)n_in; (void)out_size; (void)ws_size;
  const float* features = (const float*)d_in[0];
  const int*   labels   = (const int*)d_in[1];
  const int*   src      = (const int*)d_in[2];
  const int*   dst      = (const int*)d_in[3];
  const float* vals     = (const float*)d_in[4];
  const float* W0       = (const float*)d_in[5];
  const float* b0       = (const float*)d_in[6];
  const float* W1       = (const float*)d_in[7];
  const float* b1       = (const float*)d_in[8];
  const float* W2       = (const float*)d_in[9];
  const float* b2       = (const float*)d_in[10];
  float* out = (float*)d_out;

  char* ws = (char*)d_ws;
  size_t off = 0;
  auto alloc = [&](size_t bytes) -> void* {
    void* p = ws + off;
    off += (bytes + 255) & ~(size_t)255;
    return p;
  };
  float*    A   = (float*)alloc((size_t)Nn * 128 * sizeof(float));
  float*    B   = (float*)alloc((size_t)Nn * 128 * sizeof(float));
  float*    LA  = (float*)alloc((size_t)Nn * NCP * sizeof(float));
  float*    LB  = (float*)alloc((size_t)Nn * NCP * sizeof(float));
  _Float16* Wt0 = (_Float16*)alloc(128 * 128 * sizeof(_Float16));
  _Float16* Wt1 = (_Float16*)alloc(128 * 128 * sizeof(_Float16));
  _Float16* Wt2 = (_Float16*)alloc(NCP * 128 * sizeof(_Float16));
  float*    b2p = (float*)alloc(NCP * sizeof(float));

  const int edgeBlocks = (Ee + 31) / 32;   // 4 edges per wave, 8 waves/block
  const int gemmBlocks = 1280;

  gcn_prep_kernel<<<1, 256, 0, stream>>>(W0, W1, W2, b2, Wt0, Wt1, Wt2, b2p);
  gcn_zero_kernel<<<1, 1, 0, stream>>>(out);

  // layer 0: H = features@W0+b0 -> A,B ; B += agg(A) ; (relu fused into next)
  gcn_gemm_kernel<128, false><<<gemmBlocks, 256, 0, stream>>>(features, Wt0, b0, A, B);
  gcn_edge_kernel<128><<<edgeBlocks, 256, 0, stream>>>(src, dst, vals, A, B, Ee);
  // layer 1: H = relu(B)@W1+b1 -> A,B(in-place) ; B += agg(A)
  gcn_gemm_kernel<128, true><<<gemmBlocks, 256, 0, stream>>>(B, Wt1, b1, A, B);
  gcn_edge_kernel<128><<<edgeBlocks, 256, 0, stream>>>(src, dst, vals, A, B, Ee);
  // layer 2: logits = relu(B)@W2+b2 -> LA,LB (48-wide padded) ; LB += agg(LA)
  gcn_gemm_kernel<NCP, true><<<gemmBlocks, 256, 0, stream>>>(B, Wt2, b2p, LA, LB);
  gcn_edge_kernel<NCP><<<edgeBlocks, 256, 0, stream>>>(src, dst, vals, LA, LB, Ee);
  // loss
  gcn_loss_kernel<<<(Nn + 255) / 256, 256, 0, stream>>>(LB, labels, out, Nn);
}
